// SingleHeadAttention_6390911336885
// MI455X (gfx1250) — compile-verified, matrix operand order checked by reading, not run on hardware
//
#include <hip/hip_runtime.h>

// SingleHeadAttention for MI455X (gfx1250), full WMMA-f16 pipeline.
// B=4, S=4096, H=1024, D=64. Reference output is provably zero
// (floor(softmax/8) == 0 elementwise) but the floor is applied faithfully
// where the reference applies it; the compiler cannot fold it away.

typedef __attribute__((ext_vector_type(16))) _Float16 v16h;
typedef __attribute__((ext_vector_type(8)))  _Float16 v8h;
typedef __attribute__((ext_vector_type(8)))  float    v8f;

constexpr int BATCH  = 4;
constexpr int SEQ    = 4096;
constexpr int HIDDEN = 1024;
constexpr int HEAD   = 64;
constexpr int ROWS   = BATCH * SEQ;                 // 16384
constexpr size_t QKV_E = (size_t)ROWS * HEAD;       // f16 elems per matrix
constexpr int XP     = HIDDEN + 8;                  // LDS pitch (x tile)
constexpr int SPITCH = SEQ + 8;                     // LDS pitch (score stash)
constexpr size_t ATTN_LDS_BYTES = (size_t)16 * SPITCH * sizeof(_Float16); // 131328

// workspace layout (f16): [Q | K | Vt | Wt]
//   Q  : ROWS*HEAD row-major
//   K  : ROWS*HEAD row-major
//   Vt : per batch [HEAD][SEQ] (transposed)
//   Wt : 3 * [HEAD][HIDDEN] (transposed, converted weights)

union Frag16 {
    v16h v;
    v8h  h8[2];
    _Float16 h[16];
};

__device__ __forceinline__ v8f vzero8() {
    v8f z;
#pragma unroll
    for (int i = 0; i < 8; ++i) z[i] = 0.0f;
    return z;
}

__device__ __forceinline__ v8f wmma_f16(v16h a, v16h b, v8f c) {
    return __builtin_amdgcn_wmma_f32_16x16x32_f16(false, a, false, b, (short)0, c,
                                                  false, false);
}

// ---------------------------------------------------------------------------
// Kernel 0: convert + transpose weights to f16 Wt[n][k].  256 KB x3, trivial.
// ---------------------------------------------------------------------------
__global__ __launch_bounds__(256)
void prep_w_kernel(const float* __restrict__ Wq, const float* __restrict__ Wk,
                   const float* __restrict__ Wv, _Float16* __restrict__ wt) {
    const int m = blockIdx.x;
    const float* __restrict__ W = (m == 0) ? Wq : ((m == 1) ? Wk : Wv);
    _Float16* __restrict__ o = wt + (size_t)m * HEAD * HIDDEN;
    for (int idx = threadIdx.x; idx < HEAD * HIDDEN; idx += 256) {
        const int n = idx >> 10;            // idx / HIDDEN
        const int k = idx & (HIDDEN - 1);
        o[idx] = (_Float16)W[(size_t)k * HEAD + n];   // Wt[n][k] = W[k][n]
    }
}

// ---------------------------------------------------------------------------
// Kernel 1: QKV projection.  grid = ROWS/16 blocks, block = 96 (3 waves).
// The 3 waves share one x-tile (16x1024 f16) staged in LDS; wave w computes
// matrix w (0=Q,1=K,2=V).  A-frags via ds_load_b128, B-frags via contiguous
// 16B loads from Wt.  V is stored transposed for the PV stage.
// ---------------------------------------------------------------------------
__global__ __launch_bounds__(96)
void qkv_proj_kernel(const float* __restrict__ x,
                     const float* __restrict__ bq, const float* __restrict__ bk,
                     const float* __restrict__ bv, _Float16* __restrict__ ws) {
    __shared__ __align__(16) _Float16 xs[16 * XP];

    const int tid  = threadIdx.x;
    const int row0 = blockIdx.x * 16;

    // Cooperative load+convert: the 16 rows are contiguous in x.
    const float* __restrict__ xb = x + (size_t)row0 * HIDDEN;
    for (int p = tid; p < 16 * HIDDEN / 2; p += 96) {
        const float2 v = *(const float2*)(xb + 2 * (size_t)p);
        const int row = p >> 9;                     // p / (HIDDEN/2)
        const int c   = (p & (HIDDEN / 2 - 1)) * 2;
        xs[row * XP + c]     = (_Float16)v.x;
        xs[row * XP + c + 1] = (_Float16)v.y;
    }
    __syncthreads();

    const int w     = tid >> 5;           // wave = matrix index
    const int lane  = tid & 31;
    const int nr    = lane & 15;
    const int half8 = (lane >> 4) << 3;

    const float* __restrict__ bias = (w == 0) ? bq : ((w == 1) ? bk : bv);
    const _Float16* __restrict__ Wt = ws + 3 * QKV_E + (size_t)w * HEAD * HIDDEN;

    v8f acc[4];
#pragma unroll
    for (int t = 0; t < 4; ++t) acc[t] = vzero8();

    for (int kk = 0; kk < HIDDEN; kk += 32) {
        Frag16 a;
        a.h8[0] = *(const v8h*)(&xs[nr * XP + kk + half8]);
        a.h8[1] = *(const v8h*)(&xs[nr * XP + kk + half8 + 16]);
        Frag16 bfr[4];
#pragma unroll
        for (int t = 0; t < 4; ++t) {
            const _Float16* __restrict__ wp =
                Wt + (size_t)(16 * t + nr) * HIDDEN + kk + half8;
            bfr[t].h8[0] = *(const v8h*)(wp);
            bfr[t].h8[1] = *(const v8h*)(wp + 16);
        }
#pragma unroll
        for (int t = 0; t < 4; ++t) acc[t] = wmma_f16(a.v, bfr[t].v, acc[t]);
    }

    // C layout: lane holds col N = nr+16t, rows r + half8.
    if (w < 2) {
        _Float16* __restrict__ out = ws + (size_t)w * QKV_E;  // Q or K row-major
#pragma unroll
        for (int t = 0; t < 4; ++t) {
            const int col = 16 * t + nr;
            const float bc = bias[col];
#pragma unroll
            for (int r = 0; r < 8; ++r)
                out[(size_t)(row0 + r + half8) * HEAD + col] =
                    (_Float16)(acc[t][r] + bc);
        }
    } else {
        const int batch = row0 / SEQ;                 // 16-row block never straddles
        const int srow0 = row0 - batch * SEQ;
        _Float16* __restrict__ vt =
            ws + 2 * QKV_E + (size_t)batch * HEAD * SEQ;
#pragma unroll
        for (int t = 0; t < 4; ++t) {
            const int col = 16 * t + nr;
            const float bc = bias[col];
#pragma unroll
            for (int r = 0; r < 8; ++r)               // contiguous over r
                vt[(size_t)col * SEQ + srow0 + half8 + r] =
                    (_Float16)(acc[t][r] + bc);
        }
    }
}

// ---------------------------------------------------------------------------
// Kernel 2: causal attention.  grid = (SEQ/16, BATCH), block = 32 (1 wave).
// Score pass (WMMA) -> f16 stash in LDS -> lane-local max/sum sweeps
// (1 shuffle each) -> PV pass reading P directly in A-layout, V from Vt.
// ---------------------------------------------------------------------------
__global__ __launch_bounds__(32)
void attn_kernel(const _Float16* __restrict__ ws, float* __restrict__ out) {
    extern __shared__ __align__(16) _Float16 stash[];   // [16][SPITCH]

    const _Float16* __restrict__ Q = ws;
    const _Float16* __restrict__ K = ws + QKV_E;

    const int lane  = threadIdx.x;
    const int nr    = lane & 15;
    const int half8 = (lane >> 4) << 3;
    const int qblk  = blockIdx.x;
    const size_t bbase = (size_t)blockIdx.y * SEQ;
    const _Float16* __restrict__ VT =
        ws + 2 * QKV_E + (size_t)blockIdx.y * HEAD * SEQ;

    // Q A-fragments (d = 0..31, 32..63).
    Frag16 qa[2];
    {
        const _Float16* __restrict__ qrow =
            Q + (bbase + (size_t)qblk * 16 + nr) * HEAD;
#pragma unroll
        for (int c = 0; c < 2; ++c) {
            qa[c].h8[0] = *(const v8h*)(qrow + c * 32 + half8);
            qa[c].h8[1] = *(const v8h*)(qrow + c * 32 + half8 + 16);
        }
    }

    const int jmax = qblk;

    // ---- Score pass: S = Q Kt, causal mask, stash as f16. ----
    for (int j = 0; j <= jmax; ++j) {
        const _Float16* __restrict__ krow =
            K + (bbase + (size_t)j * 16 + nr) * HEAD;
        Frag16 kb0, kb1;
        kb0.h8[0] = *(const v8h*)(krow + half8);
        kb0.h8[1] = *(const v8h*)(krow + half8 + 16);
        kb1.h8[0] = *(const v8h*)(krow + 32 + half8);
        kb1.h8[1] = *(const v8h*)(krow + 32 + half8 + 16);
        v8f s = vzero8();
        s = wmma_f16(qa[0].v, kb0.v, s);
        s = wmma_f16(qa[1].v, kb1.v, s);
        if (j == jmax) {
            const int kj = j * 16 + nr;
#pragma unroll
            for (int r = 0; r < 8; ++r)
                if (kj > qblk * 16 + r + half8) s[r] = -1.0e30f;  // -> f16 -inf
        }
#pragma unroll
        for (int r = 0; r < 8; ++r)
            stash[(r + half8) * SPITCH + j * 16 + nr] = (_Float16)s[r];
    }
    __syncthreads();

    // A-layout sweep base: lane owns row nr, keys {j*16+half8+i, +16+i}.
    // Partner lane (lane^16) owns the other half of the same row.
    // ---- Max sweep ----
    float m_row = -1.0e30f;
    for (int j = 0; j <= jmax; j += 2) {
        const v8h e0 = *(const v8h*)(&stash[nr * SPITCH + j * 16 + half8]);
#pragma unroll
        for (int i = 0; i < 8; ++i) m_row = fmaxf(m_row, (float)e0[i]);
        if (j + 1 <= jmax) {
            const v8h e1 = *(const v8h*)(&stash[nr * SPITCH + j * 16 + half8 + 16]);
#pragma unroll
            for (int i = 0; i < 8; ++i) m_row = fmaxf(m_row, (float)e1[i]);
        }
    }
    m_row = fmaxf(m_row, __shfl_xor(m_row, 16, 32));

    // ---- Sum sweep: overwrite stash with e = exp(s - m). ----
    float l_row = 0.0f;
    for (int j = 0; j <= jmax; j += 2) {
        _Float16* p0 = &stash[nr * SPITCH + j * 16 + half8];
        v8h v0 = *(const v8h*)p0, w0;
#pragma unroll
        for (int i = 0; i < 8; ++i) {
            const float e = __expf((float)v0[i] - m_row);
            l_row += e;
            w0[i] = (_Float16)e;
        }
        *(v8h*)p0 = w0;
        if (j + 1 <= jmax) {
            _Float16* p1 = p0 + 16;
            v8h v1 = *(const v8h*)p1, w1;
#pragma unroll
            for (int i = 0; i < 8; ++i) {
                const float e = __expf((float)v1[i] - m_row);
                l_row += e;
                w1[i] = (_Float16)e;
            }
            *(v8h*)p1 = w1;
        }
    }
    l_row += __shfl_xor(l_row, 16, 32);
    const float pscale = 0.125f / l_row;     // floor((e/l)/8) = floor(e*pscale)
    __syncthreads();

    // ---- PV pass: P = floor(e * pscale) as A-frag, O += P @ V. ----
    v8f o[4];
#pragma unroll
    for (int t = 0; t < 4; ++t) o[t] = vzero8();

    for (int j = 0; j <= jmax; j += 2) {
        Frag16 pa;
        const v8h e0 = *(const v8h*)(&stash[nr * SPITCH + j * 16 + half8]);
#pragma unroll
        for (int i = 0; i < 8; ++i)
            pa.h[i] = (_Float16)floorf((float)e0[i] * pscale);
        if (j + 1 <= jmax) {
            const v8h e1 = *(const v8h*)(&stash[nr * SPITCH + j * 16 + half8 + 16]);
#pragma unroll
            for (int i = 0; i < 8; ++i)
                pa.h[8 + i] = (_Float16)floorf((float)e1[i] * pscale);
        } else {
#pragma unroll
            for (int i = 0; i < 8; ++i) pa.h[8 + i] = (_Float16)0.0f;
        }
        // V B-frags from Vt: contiguous; keys <= jmax*16+31 <= 4095 always.
#pragma unroll
        for (int t = 0; t < 4; ++t) {
            const _Float16* __restrict__ vp =
                VT + (size_t)(16 * t + nr) * SEQ + j * 16 + half8;
            Frag16 vb;
            vb.h8[0] = *(const v8h*)(vp);
            vb.h8[1] = *(const v8h*)(vp + 16);
            o[t] = wmma_f16(pa.v, vb.v, o[t]);
        }
    }

    // Store O (f32).
#pragma unroll
    for (int t = 0; t < 4; ++t) {
        const int col = 16 * t + nr;
#pragma unroll
        for (int r = 0; r < 8; ++r)
            out[(bbase + (size_t)qblk * 16 + r + half8) * HEAD + col] = o[t][r];
    }
}

extern "C" void kernel_launch(void* const* d_in, const int* in_sizes, int n_in,
                              void* d_out, int out_size, void* d_ws, size_t ws_size,
                              hipStream_t stream) {
    const float* x  = (const float*)d_in[0];
    const float* Wq = (const float*)d_in[1];
    const float* bq = (const float*)d_in[2];
    const float* Wk = (const float*)d_in[3];
    const float* bk = (const float*)d_in[4];
    const float* Wv = (const float*)d_in[5];
    const float* bv = (const float*)d_in[6];
    _Float16* ws = (_Float16*)d_ws;  // needs 3*QKV (6 MB) + Wt (384 KB)
    float* out = (float*)d_out;

    (void)hipFuncSetAttribute((const void*)attn_kernel,
                              hipFuncAttributeMaxDynamicSharedMemorySize,
                              (int)ATTN_LDS_BYTES);

    prep_w_kernel<<<dim3(3), 256, 0, stream>>>(Wq, Wk, Wv, ws + 3 * QKV_E);

    qkv_proj_kernel<<<dim3(ROWS / 16), 96, 0, stream>>>(x, bq, bk, bv, ws);

    attn_kernel<<<dim3(SEQ / 16, BATCH), 32, ATTN_LDS_BYTES, stream>>>(ws, out);
}
